// CoxNNetLoss_75368086110967
// MI455X (gfx1250) — compile-verified, matrix operand order checked by reading, not run on hardware
//
#include <hip/hip_runtime.h>
#include <math.h>

typedef __attribute__((ext_vector_type(16))) _Float16 v16h;
typedef __attribute__((ext_vector_type(8)))  _Float16 v8h;
typedef __attribute__((ext_vector_type(8)))  float    v8f;
typedef __attribute__((ext_vector_type(4)))  float    v4f;

// ---------------------------------------------------------------------------
// Kernel 1: E[j] = (f16) exp(theta[j])
// ---------------------------------------------------------------------------
__global__ void cox_exp_kernel(const float* __restrict__ theta,
                               _Float16* __restrict__ Eh, int n) {
    int i = blockIdx.x * blockDim.x + threadIdx.x;
    if (i < n) Eh[i] = (_Float16)expf(theta[i]);
}

// ---------------------------------------------------------------------------
// Kernel 2: risk[i] = sum_j [d_j >= d_i] * exp(theta_j)
//
// Each wave owns a 16-row block of i. Per K=32 tile, each lane builds its
// slice of the f16 A operand as masked exp-values; B is all-ones, so
// D[i,c] = sum_k A[i,k] for every column c (layout-invariant over K, fp32
// accumulation). LDS reads are unconditional ds_load_b128s; the mask is a
// branchless register select, so WMMA co-executes with the VALU tile build.
// ---------------------------------------------------------------------------
__global__ __launch_bounds__(256) void cox_riskset_kernel(
    const float* __restrict__ dur, const _Float16* __restrict__ Eh,
    float* __restrict__ risk, int n) {
    extern __shared__ char smem[];
    float*    s_d = (float*)smem;
    _Float16* s_e = (_Float16*)(smem + (size_t)n * sizeof(float));

    for (int t = threadIdx.x; t < n; t += blockDim.x) {
        s_d[t] = dur[t];
        s_e[t] = Eh[t];
    }
    __syncthreads();

    const int lane   = threadIdx.x & 31;
    const int waveIb = ((blockIdx.x * (blockDim.x >> 5)) + (threadIdx.x >> 5)) * 16;
    if (waveIb >= n) return;                    // wave-uniform: EXEC stays full

    const int   m   = lane & 15;                // A-matrix row M = lane % 16
    const float di  = s_d[waveIb + m];
    const int   kb0 = (lane & 16) ? 8 : 0;      // K partition per half-wave
    const int   kb1 = kb0 + 16;

    v16h bOnes;
#pragma unroll
    for (int e = 0; e < 16; ++e) bOnes[e] = (_Float16)1.0f;

    v8f acc = {};
#pragma unroll 2
    for (int jb = 0; jb < n; jb += 32) {
        // Unconditional vector loads from LDS (16B aligned: jb%32==0, kb0 in {0,8})
        const float*    pd0 = s_d + jb + kb0;
        const float*    pd1 = s_d + jb + kb1;
        const v4f dA = *(const v4f*)(pd0);
        const v4f dB = *(const v4f*)(pd0 + 4);
        const v4f dC = *(const v4f*)(pd1);
        const v4f dD = *(const v4f*)(pd1 + 4);
        const v8h eA = *(const v8h*)(s_e + jb + kb0);
        const v8h eB = *(const v8h*)(s_e + jb + kb1);

        v16h a;
#pragma unroll
        for (int e = 0; e < 4; ++e) {
            a[e]      = (dA[e] >= di) ? eA[e]     : (_Float16)0.0f;
            a[4 + e]  = (dB[e] >= di) ? eA[4 + e] : (_Float16)0.0f;
            a[8 + e]  = (dC[e] >= di) ? eB[e]     : (_Float16)0.0f;
            a[12 + e] = (dD[e] >= di) ? eB[4 + e] : (_Float16)0.0f;
        }
        acc = __builtin_amdgcn_wmma_f32_16x16x32_f16(
            /*neg_a=*/false, a, /*neg_b=*/false, bOnes,
            /*c_mod=*/(short)0, acc, /*reuse_a=*/false, /*reuse_b=*/false);
    }

    // D layout: VGPR r holds M=r (lanes 0-15) and M=r+8 (lanes 16-31);
    // all columns identical, so read column 0 at lanes 0 and 16.
    if (lane == 0) {
#pragma unroll
        for (int r = 0; r < 8; ++r) risk[waveIb + r] = acc[r];
    } else if (lane == 16) {
#pragma unroll
        for (int r = 0; r < 8; ++r) risk[waveIb + 8 + r] = acc[r];
    }
}

// ---------------------------------------------------------------------------
// Kernel 3: out = -(1/n) * sum_i (theta_i - log(risk_i)) * events_i
//                 + 0.01 * sqrt(sum(W^2))
// ---------------------------------------------------------------------------
__global__ __launch_bounds__(256) void cox_finalize_kernel(
    const float* __restrict__ theta, const float* __restrict__ events,
    const float* __restrict__ risk,  const float* __restrict__ W,
    float* __restrict__ out, int n, int wn) {
    __shared__ float r1[256];
    __shared__ float r2[256];
    float s_cox = 0.0f, s_w2 = 0.0f;
    for (int i = threadIdx.x; i < n; i += blockDim.x)
        s_cox += (theta[i] - logf(risk[i])) * events[i];
    for (int i = threadIdx.x; i < wn; i += blockDim.x) {
        float w = W[i];
        s_w2 += w * w;
    }
    r1[threadIdx.x] = s_cox;
    r2[threadIdx.x] = s_w2;
    __syncthreads();
    for (int s = 128; s > 0; s >>= 1) {
        if ((int)threadIdx.x < s) {
            r1[threadIdx.x] += r1[threadIdx.x + s];
            r2[threadIdx.x] += r2[threadIdx.x + s];
        }
        __syncthreads();
    }
    if (threadIdx.x == 0)
        out[0] = -r1[0] / (float)n + 0.01f * sqrtf(r2[0]);
}

// ---------------------------------------------------------------------------
extern "C" void kernel_launch(void* const* d_in, const int* in_sizes, int n_in,
                              void* d_out, int out_size, void* d_ws, size_t ws_size,
                              hipStream_t stream) {
    const float* hazard = (const float*)d_in[0];  // [n,1] f32 (flat = theta)
    const float* dur    = (const float*)d_in[1];  // [n]   f32
    const float* events = (const float*)d_in[2];  // [n]   f32
    const float* W      = (const float*)d_in[3];  // [512*256] f32
    const int n  = in_sizes[0];
    const int wn = in_sizes[3];
    float* out = (float*)d_out;

    // workspace: risk[n] f32, then Eh[n] f16
    float*    risk = (float*)d_ws;
    _Float16* Eh   = (_Float16*)((char*)d_ws + (size_t)n * sizeof(float));

    cox_exp_kernel<<<(n + 255) / 256, 256, 0, stream>>>(hazard, Eh, n);

    const int waves  = (n + 15) / 16;
    const int blocks = (waves + 7) / 8;           // 8 waves / 256-thread block
    const size_t smem = (size_t)n * (sizeof(float) + sizeof(_Float16)); // 96 KB
    cox_riskset_kernel<<<blocks, 256, smem, stream>>>(dur, Eh, risk, n);

    cox_finalize_kernel<<<1, 256, 0, stream>>>(hazard, events, risk, W, out, n, wn);
}
